// ContinuousConv_34411277976254
// MI455X (gfx1250) — matrix operand order repaired and checked by reading, not compiled, not run on hardware
//
#include <hip/hip_runtime.h>

typedef __attribute__((ext_vector_type(16))) _Float16 v16h;
typedef __attribute__((ext_vector_type(8)))  _Float16 v8h;
typedef __attribute__((ext_vector_type(2)))  _Float16 h2;
typedef __attribute__((ext_vector_type(8)))  float    v8f;

#define TILE 16
#define WAVES_PER_BLOCK 8
#define THREADS (WAVES_PER_BLOCK * 32)

// All tile LDS is wave-private; LDS ops from one wave complete in order,
// so a dscnt drain + compiler barriers is enough (no __syncthreads in the
// per-wave loop, which would deadlock with divergent trip counts).
__device__ __forceinline__ void wave_lds_sync() {
  __builtin_amdgcn_wave_barrier();
  asm volatile("s_wait_dscnt 0" ::: "memory");
  __builtin_amdgcn_wave_barrier();
}

__global__ void __launch_bounds__(256) cconv_zero(float* out, float* cnt,
                                                  int nOut, int nNodes) {
  int i = blockIdx.x * blockDim.x + threadIdx.x;
  if (i < nOut)   out[i] = 0.f;
  if (i < nNodes) cnt[i] = 0.f;
}

__global__ void __launch_bounds__(THREADS) cconv_main(
    const float* __restrict__ pos, const float* __restrict__ feat,
    const float* __restrict__ filt, const int* __restrict__ ei,
    float* __restrict__ sums, float* __restrict__ cnt,
    int nEdges, int nTiles)
{
  // Filter bank, f16, transposed to [cell][cout][cin] so a B-fragment column
  // (K = 16 cin of one cell, one cout) is 32 contiguous bytes.
  __shared__ _Float16 sFt[64 * 16 * 16];                 // 32 KB
  __shared__ _Float16 sW [WAVES_PER_BLOCK][16 * 64];     // per-wave weight tile
  __shared__ _Float16 sF [WAVES_PER_BLOCK][16 * 16];     // per-wave feature tile
  __shared__ int      sRow[WAVES_PER_BLOCK][16];
  __shared__ float    sWin[WAVES_PER_BLOCK][16];

  for (int t = threadIdx.x; t < 64 * 256; t += THREADS) {
    int g = t >> 8;            // cell
    int i = (t >> 4) & 15;     // cin
    int o = t & 15;            // cout
    sFt[g * 256 + o * 16 + i] = (_Float16)filt[t];
  }
  __syncthreads();

  const int lane = threadIdx.x & 31;
  const int wv   = threadIdx.x >> 5;
  const int nW   = gridDim.x * WAVES_PER_BLOCK;
  _Float16* W   = sW[wv];
  _Float16* F   = sF[wv];
  int*      RW  = sRow[wv];
  float*    WIN = sWin[wv];

  const int eA = lane & 15;   // A-matrix row (edge in tile)
  const int kh = lane >> 4;   // K half (A/B lane-half per ISA layouts)
  const int n  = lane & 15;   // output column (cout)

  for (int tile = blockIdx.x * WAVES_PER_BLOCK + wv; tile < nTiles; tile += nW) {
    // ---- zero the 16x64 f16 weight tile (512 dwords) ----
    unsigned int* wz = (unsigned int*)W;
    #pragma unroll
    for (int j = 0; j < 16; ++j) wz[lane + 32 * j] = 0u;
    wave_lds_sync();

    {
      int nxt = tile + nW;
      if (nxt < nTiles && lane == 0) __builtin_prefetch(&ei[nxt * TILE], 0, 3);
    }

    // ---- per-edge scalars: one edge per lane 0..15 ----
    if (lane < 16) {
      int e = tile * TILE + lane;
      int row = -1;
      float window = 0.f;
      if (e < nEdges) {
        row      = ei[e];
        int col  = ei[nEdges + e];
        float rx = pos[col * 3 + 0] - pos[row * 3 + 0];
        float ry = pos[col * 3 + 1] - pos[row * 3 + 1];
        float rz = pos[col * 3 + 2] - pos[row * 3 + 2];
        float d2 = rx * rx + ry * ry + rz * rz;
        float t1 = 1.f - d2 * 4.f;                 // 1 - d2/R^2, R=0.5
        window   = (d2 < 0.25f) ? t1 * t1 * t1 : 0.f;
        float nrm = sqrtf(d2);
        float s   = tanhf(nrm) / (nrm + 1e-8f);
        float a   = (rz * s + 1.f) * 1.5f;         // gc[:,2]
        float b   = (ry * s + 1.f) * 1.5f;         // gc[:,1]
        float c   = (rx * s + 1.f) * 1.5f;         // gc[:,0]
        float a0f = floorf(a), b0f = floorf(b), c0f = floorf(c);
        int a0 = min(max((int)a0f, 0), 3);
        int b0 = min(max((int)b0f, 0), 3);
        int c0 = min(max((int)c0f, 0), 3);
        int a1 = min(a0 + 1, 3), b1 = min(b0 + 1, 3), c1 = min(c0 + 1, 3);
        float ad = a - a0f, bd = b - b0f, cd = c - c0f;
        float wa[2] = {1.f - ad, ad};
        float wb[2] = {1.f - bd, bd};
        float wc[2] = {1.f - cd, cd};
        int   ia[2] = {a0, a1}, jb[2] = {b0, b1}, kc[2] = {c0, c1};
        _Float16* Wr = &W[lane * 64];
        #pragma unroll
        for (int da = 0; da < 2; ++da)
          #pragma unroll
          for (int db = 0; db < 2; ++db)
            #pragma unroll
            for (int dc = 0; dc < 2; ++dc) {
              int cell = ia[da] * 16 + jb[db] * 4 + kc[dc];
              // accumulate (handles clamp-duplicated corners); in-wave LDS
              // RMW is in-order
              Wr[cell] = Wr[cell] + (_Float16)(wa[da] * wb[db] * wc[dc]);
            }
        const float* fp = &feat[col * 16];
        #pragma unroll
        for (int i = 0; i < 16; ++i) F[lane * 16 + i] = (_Float16)fp[i];
      } else {
        #pragma unroll
        for (int i = 0; i < 16; ++i) F[lane * 16 + i] = (_Float16)0.f;
      }
      RW[lane]  = row;
      WIN[lane] = window;
    }
    wave_lds_sync();

    // Per-lane feature slice: cin = kh*8 .. kh*8+7 of edge eA (16B aligned)
    v8h fv = *(const v8h*)&F[eA * 16 + kh * 8];

    // ---- GEMM: D[16e x 16o] += A[16e x 1024] * B[1024 x 16o], K-step 32 ----
    // Step s covers cells g0=2s (K 0..15 of step) and g1=2s+1 (K 16..31).
    // A layout (16-bit 16x32): lane half kh holds K = kh*8+j (lo VGPRs) and
    // 16+kh*8+j (hi VGPRs)  =>  A[j] = W[e][2s]*f[kh*8+j], A[8+j] = W[e][2s+1]*f[kh*8+j].
    v8f acc = {};
    #pragma unroll 8
    for (int s = 0; s < 32; ++s) {
      h2 w01 = *(const h2*)&W[eA * 64 + 2 * s];   // packed (w[2s], w[2s+1])
      v16h A;
      #pragma unroll
      for (int j = 0; j < 8; ++j) {
        A[j]     = w01[0] * fv[j];
        A[8 + j] = w01[1] * fv[j];
      }
      // B (16-bit 32x16): lane half kh holds K = kh*8+j (lo) and 16+kh*8+j (hi) of column n
      v8h b0 = *(const v8h*)&sFt[(2 * s) * 256 + n * 16 + kh * 8];
      v8h b1 = *(const v8h*)&sFt[(2 * s + 1) * 256 + n * 16 + kh * 8];
      v16h B = __builtin_shufflevector(b0, b1, 0, 1, 2, 3, 4, 5, 6, 7, 8, 9, 10, 11, 12, 13, 14, 15);
      acc = __builtin_amdgcn_wmma_f32_16x16x32_f16(
          false, A, false, B, (short)0, acc, false, false);
    }

    // ---- window scale + segment sum (C/D layout: VGPR r -> edge r + 8*kh) ----
    #pragma unroll
    for (int r = 0; r < 8; ++r) {
      int m   = r + 8 * kh;
      int row = RW[m];
      if (row >= 0) atomicAdd(&sums[row * 16 + n], acc[r] * WIN[m]);
    }
    if (lane < 16) {
      int row = RW[lane];
      if (row >= 0) atomicAdd(&cnt[row], 1.0f);
    }
    wave_lds_sync();
  }
}

__global__ void __launch_bounds__(256) cconv_div(float* out, const float* cnt,
                                                 int nNodes) {
  int i = blockIdx.x * blockDim.x + threadIdx.x;
  if (i < nNodes * 16) out[i] = out[i] / fmaxf(cnt[i >> 4], 1.f);
}

extern "C" void kernel_launch(void* const* d_in, const int* in_sizes, int n_in,
                              void* d_out, int out_size, void* d_ws, size_t ws_size,
                              hipStream_t stream) {
  const float* pos  = (const float*)d_in[0];
  const float* feat = (const float*)d_in[1];
  const float* filt = (const float*)d_in[2];
  const int*   ei   = (const int*)d_in[3];
  const int nNodes = in_sizes[0] / 3;
  const int nEdges = in_sizes[3] / 2;
  const int nTiles = (nEdges + TILE - 1) / TILE;

  float* sums = (float*)d_out;   // accumulate directly into output
  float* cnt  = (float*)d_ws;    // nNodes floats of scratch

  int zBlocks = (nNodes * 16 + 255) / 256;
  cconv_zero<<<zBlocks, 256, 0, stream>>>(sums, cnt, nNodes * 16, nNodes);

  int blocks = 512;
  if (blocks > nTiles) blocks = nTiles > 0 ? nTiles : 1;
  cconv_main<<<blocks, THREADS, 0, stream>>>(pos, feat, filt, ei, sums, cnt,
                                             nEdges, nTiles);

  cconv_div<<<(nNodes * 16 + 255) / 256, 256, 0, stream>>>(sums, cnt, nNodes);
}
